// RSSM_Delta_45363444580943
// MI455X (gfx1250) — compile-verified, matrix operand order checked by reading, not run on hardware
//
#include <hip/hip_runtime.h>
#include <hip/hip_bf16.h>
#include <cstdint>
#include <cstddef>

typedef __attribute__((ext_vector_type(16))) __bf16 v16bf;
typedef __attribute__((ext_vector_type(8)))  __bf16 v8bf;
typedef __attribute__((ext_vector_type(8)))  float  v8f;
typedef __attribute__((ext_vector_type(4)))  unsigned int v4ui;
typedef __attribute__((ext_vector_type(4)))  int v4i;

#define BB 64
#define TT 256
#define II 512
#define SS 256
#define DD 1024
#define HH 1024

// ---------------------------------------------------------------------------
// WMMA fragment load (works for global or LDS generic pointers):
// lane L covers row (L&15); K-chunks at 8*(L>>4) and 16+8*(L>>4) per the
// CDNA5 16-bit A/B VGPR layout (two 16B vector loads).
// ---------------------------------------------------------------------------
__device__ __forceinline__ v16bf load_frag(const __bf16* p) {
  v8bf lo = *(const v8bf*)(p);
  v8bf hi = *(const v8bf*)(p + 16);
  v16bf r;
#pragma unroll
  for (int i = 0; i < 8; ++i) { r[i] = lo[i]; r[i + 8] = hi[i]; }
  return r;
}

// ---- CDNA5 async global->LDS copy (16B), with portable fallback -----------
// Builtin signature (from hipcc diagnostic): (int4 __device__*, int4 __shared__*,
// imm offset, imm cpol); HIP allows implicit generic->AS conversions, so plain
// v4i* casts with the right pointee type are what Sema wants.
__device__ __forceinline__ void async_copy16(const __bf16* g, __bf16* l) {
#if __has_builtin(__builtin_amdgcn_global_load_async_to_lds_b128)
  __builtin_amdgcn_global_load_async_to_lds_b128((v4i*)g, (v4i*)l, 0, 0);
#else
  *(v4ui*)l = *(const v4ui*)g;     // global_load_b128 + ds_store_b128
#endif
}

__device__ __forceinline__ void wait_async_0() {
#if __has_builtin(__builtin_amdgcn_s_wait_asynccnt)
  __builtin_amdgcn_s_wait_asynccnt(0);
#else
  asm volatile("s_wait_asynccnt 0x0" ::: "memory");
#endif
}
__device__ __forceinline__ void wait_async_4() {
#if __has_builtin(__builtin_amdgcn_s_wait_asynccnt)
  __builtin_amdgcn_s_wait_asynccnt(4);
#else
  asm volatile("s_wait_asynccnt 0x4" ::: "memory");
#endif
}

// ---------------------------------------------------------------------------
// Small-M GEMM (latency path for the serial recurrence, M == 64).
// C(MxN) = A(MxK,row,bf16) @ B(NxK,row,bf16)^T [+Cin][+bias]
// epi==0: f32->Cout ; epi==1: ELU->bf16 Dout.  Direct global loads; all
// operands are L2/WGP$-resident (weights ~17MB bf16 << 192MB L2).
// ---------------------------------------------------------------------------
__global__ __launch_bounds__(256) void wmma_gemm_bf16(
    const __bf16* __restrict__ A, int lda,
    const __bf16* __restrict__ Bm, int ldb,
    float* __restrict__ Cout, __bf16* __restrict__ Dout, int ldc,
    const float* __restrict__ Cin, int ldcin,
    const float* __restrict__ bias, int K, int epi)
{
  const int w    = (int)threadIdx.x >> 5;
  const int lane = (int)threadIdx.x & 31;
  const int half = lane >> 4;
  const int lr   = lane & 15;
  const int m0 = (int)blockIdx.y * 64 + (w & 3) * 16;
  const int n0 = (int)blockIdx.x * 64 + (w >> 2) * 32;

  const __bf16* Ar  = A  + (size_t)(m0 + lr)      * (size_t)lda;
  const __bf16* Br0 = Bm + (size_t)(n0 + lr)      * (size_t)ldb;
  const __bf16* Br1 = Bm + (size_t)(n0 + 16 + lr) * (size_t)ldb;

  v8f acc0 = {}; v8f acc1 = {};
  for (int k = 0; k < K; k += 32) {
    const int ko = k + half * 8;
    v16bf a  = load_frag(Ar  + ko);
    v16bf b0 = load_frag(Br0 + ko);
    v16bf b1 = load_frag(Br1 + ko);
    acc0 = __builtin_amdgcn_wmma_f32_16x16x32_bf16(false, a, false, b0,
                                                   (short)0, acc0, false, false);
    acc1 = __builtin_amdgcn_wmma_f32_16x16x32_bf16(false, a, false, b1,
                                                   (short)0, acc1, false, false);
  }

  const int c0 = n0 + lr;
  const int c1 = c0 + 16;
  const int mr = m0 + half * 8;
  const float bia0 = bias ? bias[c0] : 0.f;
  const float bia1 = bias ? bias[c1] : 0.f;
#pragma unroll
  for (int r = 0; r < 8; ++r) {
    const int row = mr + r;
    float v0 = acc0[r] + bia0;
    float v1 = acc1[r] + bia1;
    if (Cin) {
      v0 += Cin[(size_t)row * ldcin + c0];
      v1 += Cin[(size_t)row * ldcin + c1];
    }
    if (epi == 1) {
      v0 = v0 > 0.f ? v0 : (__expf(v0) - 1.f);
      v1 = v1 > 0.f ? v1 : (__expf(v1) - 1.f);
      Dout[(size_t)row * ldc + c0] = (__bf16)v0;
      Dout[(size_t)row * ldc + c1] = (__bf16)v1;
    } else {
      Cout[(size_t)row * ldc + c0] = v0;
      Cout[(size_t)row * ldc + c1] = v1;
    }
  }
}

// ---------------------------------------------------------------------------
// Big-M GEMM (throughput path: decoder + hoisted posterior-x GEMM).
// 128x128 block tile, double-buffered async global->LDS staging (ASYNCcnt),
// 8 waves x (32M x 64N) register tiles -> 8 WMMAs per 6 LDS fragment reads.
// Requires M%128==0, N%128==0, K%32==0.
// ---------------------------------------------------------------------------
#define KT 32
#define LS 40   // padded LDS row stride (elements): 80B -> spread banks

__global__ __launch_bounds__(256) void wmma_gemm_big(
    const __bf16* __restrict__ A, int lda,
    const __bf16* __restrict__ Bm, int ldb,
    float* __restrict__ Cout, __bf16* __restrict__ Dout, int ldc,
    const float* __restrict__ bias, int K, int epi)
{
  __shared__ __bf16 As[2][128 * LS];
  __shared__ __bf16 Bs[2][128 * LS];

  const int tid  = (int)threadIdx.x;
  const int lane = tid & 31;
  const int w    = tid >> 5;
  const int half = lane >> 4;
  const int lr   = lane & 15;
  const int wm   = w & 3;                 // 32-row M strip
  const int wn   = w >> 2;                // 64-col N strip
  const size_t mBase = (size_t)blockIdx.y * 128;
  const size_t nBase = (size_t)blockIdx.x * 128;

  // staging: 128x32 bf16 tile = 512 x 16B chunks; 2 chunks/thread/tile.
  const int r0  = tid >> 2;               // rows r0 and r0+64
  const int kc0 = (tid & 3) * 8;          // element col within k-tile
  const __bf16* gA0 = A  + (mBase + r0)      * (size_t)lda + kc0;
  const __bf16* gA1 = A  + (mBase + r0 + 64) * (size_t)lda + kc0;
  const __bf16* gB0 = Bm + (nBase + r0)      * (size_t)ldb + kc0;
  const __bf16* gB1 = Bm + (nBase + r0 + 64) * (size_t)ldb + kc0;
  const int lofs0 = r0 * LS + kc0;
  const int lofs1 = (r0 + 64) * LS + kc0;

  v8f acc[2][4];
#pragma unroll
  for (int i = 0; i < 2; ++i)
#pragma unroll
    for (int j = 0; j < 4; ++j) { v8f z = {}; acc[i][j] = z; }

  const int KB = K / KT;

  // prologue: tile 0 -> buffer 0
  async_copy16(gA0, &As[0][lofs0]);
  async_copy16(gA1, &As[0][lofs1]);
  async_copy16(gB0, &Bs[0][lofs0]);
  async_copy16(gB1, &Bs[0][lofs1]);

  for (int kb = 0; kb < KB; ++kb) {
    const int cb = kb & 1;
    if (kb + 1 < KB) {                    // prefetch next tile into other buf
      const int k0 = (kb + 1) * KT;
      const int nb = cb ^ 1;
      async_copy16(gA0 + k0, &As[nb][lofs0]);
      async_copy16(gA1 + k0, &As[nb][lofs1]);
      async_copy16(gB0 + k0, &Bs[nb][lofs0]);
      async_copy16(gB1 + k0, &Bs[nb][lofs1]);
      wait_async_4();                     // tile kb's 4 copies done (in-order)
    } else {
      wait_async_0();
    }
    __syncthreads();                      // all waves' copies for cb visible

    v16bf af[2], bfr[4];
#pragma unroll
    for (int i = 0; i < 2; ++i)
      af[i] = load_frag(&As[cb][(32 * wm + 16 * i + lr) * LS + 8 * half]);
#pragma unroll
    for (int j = 0; j < 4; ++j)
      bfr[j] = load_frag(&Bs[cb][(64 * wn + 16 * j + lr) * LS + 8 * half]);
#pragma unroll
    for (int i = 0; i < 2; ++i)
#pragma unroll
      for (int j = 0; j < 4; ++j)
        acc[i][j] = __builtin_amdgcn_wmma_f32_16x16x32_bf16(
            false, af[i], false, bfr[j], (short)0, acc[i][j], false, false);

    __syncthreads();                      // cb fully consumed before reuse
  }

  // epilogue
#pragma unroll
  for (int i = 0; i < 2; ++i) {
#pragma unroll
    for (int j = 0; j < 4; ++j) {
      const int col = (int)nBase + 64 * wn + 16 * j + lr;
      const float bia = bias ? bias[col] : 0.f;
      const size_t rbase = mBase + 32 * wm + 16 * i + 8 * half;
#pragma unroll
      for (int r = 0; r < 8; ++r) {
        float v = acc[i][j][r] + bia;
        if (epi == 1) {
          v = v > 0.f ? v : (__expf(v) - 1.f);
          Dout[(rbase + r) * ldc + col] = (__bf16)v;
        } else {
          Cout[(rbase + r) * ldc + col] = v;
        }
      }
    }
  }
}

// ---------------------------------------------------------------------------
// Elementwise / packing kernels
// ---------------------------------------------------------------------------
__global__ void init_state_k(__bf16* state, float* hfp) {
  int idx = blockIdx.x * 256 + threadIdx.x;
  if (idx < BB * 1280) state[idx] = (__bf16)0.f;
  if (idx < BB * DD)   hfp[idx]   = 0.f;
}

// gates layout: [r | z | i_n | h_n] each 1024 wide (N=4096 fused GEMM)
__global__ void gru_gate_k(const float* __restrict__ gates,
                           float* __restrict__ hfp,
                           __bf16* __restrict__ state,    // [z(256) | h(1024)]
                           __bf16* __restrict__ rssm,     // (T*B, 1280)
                           int t) {
  int idx = blockIdx.x * 256 + threadIdx.x;
  if (idx >= BB * DD) return;
  int b = idx >> 10, d = idx & 1023;
  const float* g = gates + (size_t)b * 4096;
  float r  = 1.f / (1.f + __expf(-g[d]));
  float zg = 1.f / (1.f + __expf(-g[1024 + d]));
  float n  = tanhf(g[2048 + d] + r * g[3072 + d]);
  float h  = (1.f - zg) * n + zg * hfp[idx];
  hfp[idx] = h;
  state[(size_t)b * 1280 + 256 + d] = (__bf16)h;
  rssm[((size_t)t * BB + b) * 1280 + d] = (__bf16)h;
}

// mv layout per row: [mean(256) | logvar(256)]
__global__ void reparam_k(const float* __restrict__ mv,
                          const float* __restrict__ noise,   // (T,B,S)
                          __bf16* __restrict__ state,
                          __bf16* __restrict__ rssm,
                          int t) {
  int idx = blockIdx.x * 256 + threadIdx.x;
  if (idx >= BB * SS) return;
  int b = idx >> 8, s = idx & 255;
  const float* m = mv + (size_t)b * 512;
  float z = noise[((size_t)t * BB + b) * SS + s] * __expf(0.5f * m[256 + s]) + m[s];
  state[(size_t)b * 1280 + s] = (__bf16)z;
  rssm[((size_t)t * BB + b) * 1280 + DD + s] = (__bf16)z;
}

// Wg (4096 x 1280): rows 0-1023 r:[Wih_r|Whh_r], 1024-2047 z:[Wih_z|Whh_z],
// 2048-3071 i_n:[Wih_n|0], 3072-4095 h_n:[0|Whh_n]  (zero pad == free)
__global__ void pack_gru_w_k(const float* __restrict__ Wih,
                             const float* __restrict__ Whh,
                             __bf16* __restrict__ Wg) {
  int idx = blockIdx.x * 256 + threadIdx.x;
  if (idx >= 4096 * 1280) return;
  int row = idx / 1280, k = idx - row * 1280;
  float v;
  if (row < 2048)      v = (k < 256) ? Wih[(size_t)row * SS + k]
                                     : Whh[(size_t)row * DD + (k - 256)];
  else if (row < 3072) v = (k < 256) ? Wih[(size_t)row * SS + k] : 0.f;
  else                 v = (k < 256) ? 0.f
                                     : Whh[(size_t)(row - 1024) * DD + (k - 256)];
  Wg[idx] = (__bf16)v;
}

__global__ void pack_gru_b_k(const float* bih, const float* bhh, float* bg) {
  int i = blockIdx.x * 256 + threadIdx.x;
  if (i >= 4096) return;
  if (i < 2048)      bg[i] = bih[i] + bhh[i];
  else if (i < 3072) bg[i] = bih[i];
  else               bg[i] = bhh[i - 1024];
}

// Wmv (512 x 1024): h-columns of Wm (rows 0-255) and Wv (rows 256-511)
__global__ void pack_post_w_k(const float* Wm, const float* Wv, __bf16* W) {
  int idx = blockIdx.x * 256 + threadIdx.x;
  if (idx >= 512 * 1024) return;
  int row = idx >> 10, k = idx & 1023;
  float v = (row < 256) ? Wm[(size_t)row * 1536 + 512 + k]
                        : Wv[(size_t)(row - 256) * 1536 + 512 + k];
  W[idx] = (__bf16)v;
}

__global__ void pack_post_b_k(const float* bm, const float* bv, float* b) {
  int i = blockIdx.x * 256 + threadIdx.x;
  if (i >= 512) return;
  b[i] = (i < 256) ? bm[i] : bv[i - 256];
}

// Wx (512 x 512): x-columns of Wm / Wv (for the hoisted big GEMM)
__global__ void pack_x_w_k(const float* Wm, const float* Wv, __bf16* W) {
  int idx = blockIdx.x * 256 + threadIdx.x;
  if (idx >= 512 * 512) return;
  int row = idx >> 9, k = idx & 511;
  float v = (row < 256) ? Wm[(size_t)row * 1536 + k]
                        : Wv[(size_t)(row - 256) * 1536 + k];
  W[idx] = (__bf16)v;
}

// x (B,T,I) f32 -> x_bf (T*B, I) bf16  (time-major rows for the scan order)
__global__ void conv_x_k(const float* __restrict__ x, __bf16* __restrict__ xb) {
  int idx = blockIdx.x * 256 + threadIdx.x;
  if (idx >= TT * BB * II) return;
  int row = idx >> 9, i = idx & 511;
  int t = row >> 6, b = row & 63;
  xb[idx] = (__bf16)x[((size_t)b * TT + t) * II + i];
}

__global__ void conv_w_k(const float* __restrict__ s, __bf16* __restrict__ d, int n) {
  int i = blockIdx.x * 256 + threadIdx.x;
  if (i < n) d[i] = (__bf16)s[i];
}

// dist (T*B, I) + x (B,T,I) -> delta (B,T-1,I), x_hat (B,T,I)
__global__ void assemble_k(const float* __restrict__ x,
                           const float* __restrict__ dist,
                           float* __restrict__ out_delta,
                           float* __restrict__ out_xhat) {
  int idx = blockIdx.x * 256 + threadIdx.x;
  if (idx >= BB * TT * II) return;
  int b = idx >> 17;            // 256*512 = 1<<17
  int rem = idx & 131071;
  int t = rem >> 9, i = rem & 511;
  float xh;
  if (t == 0) xh = x[idx];
  else xh = x[(size_t)b * 131072 + (size_t)(t - 1) * 512 + i] +
            dist[((size_t)(t - 1) * BB + b) * 512 + i];
  out_xhat[idx] = xh;
  if (t < TT - 1)
    out_delta[((size_t)b * (TT - 1) + t) * 512 + i] =
        dist[((size_t)t * BB + b) * 512 + i];
}

// ---------------------------------------------------------------------------
// Host side
// ---------------------------------------------------------------------------
static inline char* wsa(char*& p, size_t bytes) {
  char* r = p;
  p += (bytes + 255) & ~(size_t)255;
  return r;
}

static void gemm(hipStream_t s, const __bf16* A, int lda, const __bf16* B, int ldb,
                 float* Cout, __bf16* Dout, int ldc, const float* Cin, int ldcin,
                 const float* bias, int M, int N, int K, int epi) {
  dim3 g(N / 64, M / 64, 1);
  wmma_gemm_bf16<<<g, 256, 0, s>>>(A, lda, B, ldb, Cout, Dout, ldc, Cin, ldcin,
                                   bias, K, epi);
}

static void gemm_big(hipStream_t s, const __bf16* A, int lda, const __bf16* B, int ldb,
                     float* Cout, __bf16* Dout, int ldc, const float* bias,
                     int M, int N, int K, int epi) {
  dim3 g(N / 128, M / 128, 1);
  wmma_gemm_big<<<g, 256, 0, s>>>(A, lda, B, ldb, Cout, Dout, ldc, bias, K, epi);
}

static inline int cdiv(int a, int b) { return (a + b - 1) / b; }

extern "C" void kernel_launch(void* const* d_in, const int* in_sizes, int n_in,
                              void* d_out, int out_size, void* d_ws, size_t ws_size,
                              hipStream_t stream) {
  (void)in_sizes; (void)n_in; (void)out_size; (void)ws_size;
  const float* x     = (const float*)d_in[0];
  const float* noise = (const float*)d_in[1];
  const float* W_ih  = (const float*)d_in[2];
  const float* W_hh  = (const float*)d_in[3];
  const float* b_ih  = (const float*)d_in[4];
  const float* b_hh  = (const float*)d_in[5];
  const float* Wm    = (const float*)d_in[6];
  const float* bm    = (const float*)d_in[7];
  const float* Wv    = (const float*)d_in[8];
  const float* bv    = (const float*)d_in[9];
  const float* dW0   = (const float*)d_in[10];
  const float* db0   = (const float*)d_in[11];
  const float* dW1   = (const float*)d_in[12];
  const float* db1   = (const float*)d_in[13];
  const float* dW2   = (const float*)d_in[14];
  const float* db2   = (const float*)d_in[15];
  const float* dW3   = (const float*)d_in[16];
  const float* db3   = (const float*)d_in[17];

  char* p = (char*)d_ws;
  __bf16* Wg      = (__bf16*)wsa(p, (size_t)4096 * 1280 * 2);
  float*  bg      = (float*) wsa(p, 4096 * 4);
  __bf16* Wmv     = (__bf16*)wsa(p, (size_t)512 * 1024 * 2);
  float*  bmv     = (float*) wsa(p, 512 * 4);
  __bf16* Wx      = (__bf16*)wsa(p, (size_t)512 * 512 * 2);
  __bf16* W0b     = (__bf16*)wsa(p, (size_t)HH * 1280 * 2);
  __bf16* W1b     = (__bf16*)wsa(p, (size_t)HH * HH * 2);
  __bf16* W2b     = (__bf16*)wsa(p, (size_t)HH * HH * 2);
  __bf16* W3b     = (__bf16*)wsa(p, (size_t)II * HH * 2);
  __bf16* x_bf    = (__bf16*)wsa(p, (size_t)TT * BB * II * 2);
  float*  mvx     = (float*) wsa(p, (size_t)TT * BB * 512 * 4);
  __bf16* rssm    = (__bf16*)wsa(p, (size_t)TT * BB * 1280 * 2);
  __bf16* actA    = (__bf16*)wsa(p, (size_t)TT * BB * HH * 2);
  __bf16* actB    = (__bf16*)wsa(p, (size_t)TT * BB * HH * 2);
  float*  dist    = (float*) wsa(p, (size_t)TT * BB * II * 4);
  __bf16* state   = (__bf16*)wsa(p, (size_t)BB * 1280 * 2);
  float*  h_f32   = (float*) wsa(p, (size_t)BB * DD * 4);
  float*  gates   = (float*) wsa(p, (size_t)BB * 4096 * 4);
  float*  mv      = (float*) wsa(p, (size_t)BB * 512 * 4);

  // ---- one-time packing (per call; deterministic) ----
  init_state_k<<<cdiv(BB * 1280, 256), 256, 0, stream>>>(state, h_f32);
  pack_gru_w_k<<<cdiv(4096 * 1280, 256), 256, 0, stream>>>(W_ih, W_hh, Wg);
  pack_gru_b_k<<<cdiv(4096, 256), 256, 0, stream>>>(b_ih, b_hh, bg);
  pack_post_w_k<<<cdiv(512 * 1024, 256), 256, 0, stream>>>(Wm, Wv, Wmv);
  pack_post_b_k<<<cdiv(512, 256), 256, 0, stream>>>(bm, bv, bmv);
  pack_x_w_k<<<cdiv(512 * 512, 256), 256, 0, stream>>>(Wm, Wv, Wx);
  conv_w_k<<<cdiv(HH * 1280, 256), 256, 0, stream>>>(dW0, W0b, HH * 1280);
  conv_w_k<<<cdiv(HH * HH, 256), 256, 0, stream>>>(dW1, W1b, HH * HH);
  conv_w_k<<<cdiv(HH * HH, 256), 256, 0, stream>>>(dW2, W2b, HH * HH);
  conv_w_k<<<cdiv(II * HH, 256), 256, 0, stream>>>(dW3, W3b, II * HH);
  conv_x_k<<<cdiv(TT * BB * II, 256), 256, 0, stream>>>(x, x_bf);

  // ---- hoisted x-half of the posterior: mvx = X @ [Wm_x;Wv_x]^T ----
  gemm_big(stream, x_bf, II, Wx, II, mvx, nullptr, 512, nullptr,
           TT * BB, 512, II, 0);

  // ---- serial recurrence: 256 steps x (2 WMMA GEMMs + 2 tiny kernels) ----
  for (int t = 0; t < TT; ++t) {
    // gates(64x4096) = [z|h](64x1280) @ Wg^T + bg
    gemm(stream, state, 1280, Wg, 1280, gates, nullptr, 4096, nullptr, 0, bg,
         BB, 4096, 1280, 0);
    gru_gate_k<<<cdiv(BB * DD, 256), 256, 0, stream>>>(gates, h_f32, state,
                                                       rssm, t);
    // mv(64x512) = h(64x1024) @ Wmv^T + mvx[t] + bmv
    gemm(stream, state + 256, 1280, Wmv, 1024, mv, nullptr, 512,
         mvx + (size_t)t * BB * 512, 512, bmv, BB, 512, DD, 0);
    reparam_k<<<cdiv(BB * SS, 256), 256, 0, stream>>>(mv, noise, state, rssm, t);
  }

  // ---- decoder: 4 big WMMA GEMMs (LDS/async staged; bias+ELU fused) ----
  gemm_big(stream, rssm, 1280, W0b, 1280, nullptr, actA, HH, db0,
           TT * BB, HH, 1280, 1);
  gemm_big(stream, actA, HH, W1b, HH, nullptr, actB, HH, db1,
           TT * BB, HH, HH, 1);
  gemm_big(stream, actB, HH, W2b, HH, nullptr, actA, HH, db2,
           TT * BB, HH, HH, 1);
  gemm_big(stream, actA, HH, W3b, HH, dist, nullptr, II, db3,
           TT * BB, II, HH, 0);

  // ---- outputs: delta (B,T-1,I) then x_hat (B,T,I), flat-concatenated ----
  float* out_delta = (float*)d_out;
  float* out_xhat  = out_delta + (size_t)BB * (TT - 1) * II;
  assemble_k<<<cdiv(BB * TT * II, 256), 256, 0, stream>>>(x, dist, out_delta,
                                                          out_xhat);
}
